// DiagnosticHRRPhasor_48232482734967
// MI455X (gfx1250) — compile-verified, matrix operand order checked by reading, not run on hardware
//
#include <hip/hip_runtime.h>
#include <hip/hip_bf16.h>
#include <math.h>

// ---------------------------------------------------------------------------
// Problem constants (fixed by the reference: B=4, S=4096, DIM=1024, CHUNK=64)
// ---------------------------------------------------------------------------
#define DIMN   1024
#define SEQ    4096
#define BATCH  4
#define MROWS  (BATCH * SEQ)     // 16384
#define CHUNKN 64

typedef __attribute__((ext_vector_type(16))) __bf16 v16bf;
typedef __attribute__((ext_vector_type(8)))  float  v8f;

union Frag {
    uint4 q[2];
    v16bf v;
};

__device__ __forceinline__ unsigned short f2bf(float f) {
    // round-to-nearest-even fp32 -> bf16
    unsigned u = __float_as_uint(f);
    unsigned r = u + 0x7FFFu + ((u >> 16) & 1u);
    return (unsigned short)(r >> 16);
}

// ---------------------------------------------------------------------------
// fp32 -> bf16 convert
// ---------------------------------------------------------------------------
__global__ __launch_bounds__(256)
void k_f32_to_bf16(const float* __restrict__ in, unsigned short* __restrict__ out, int n) {
    int i = blockIdx.x * 256 + threadIdx.x;
    if (i < n) out[i] = f2bf(in[i]);
}

// ---------------------------------------------------------------------------
// WMMA GEMM: out[m,n] = sum_k A[m,k] * W[n,k]  (+ bias[n]) with fused epilogue
//   A: MROWS x DIMN bf16 (row major), W: DIMN x DIMN bf16 (row major, N x K)
// Block tile 256(M) x 128(N), 256 threads = 8 waves as 4(M) x 2(N).
// Wave tile 64(M) x 64(N) = 4 x 4 WMMA accumulators (128 acc VGPRs):
// 16 WMMAs per 8KB of fragment loads -> 512 B per 16KFLOP WMMA, which keeps
// the XDL pipes fed from WGP$/L2 (weights are 2MB, L2-resident on 192MB L2).
// MODE: 0 = f32 store (+bias)           -> outF
//       1 = bf16 store (+bias)          -> outBF
//       2 = tanh(acc+bias)*aux[col]     -> outF   (phase projection)
//       3 = acc+bias+aux[m*DIMN+n]      -> outF   (residual output proj)
// ---------------------------------------------------------------------------
template<int MODE>
__global__ __launch_bounds__(256)
void k_gemm(const unsigned short* __restrict__ A,
            const unsigned short* __restrict__ W,
            const float* __restrict__ bias,
            const float* __restrict__ aux,
            float* __restrict__ outF,
            unsigned short* __restrict__ outBF)
{
    const int tid  = threadIdx.x;
    const int wave = tid >> 5;
    const int lane = tid & 31;
    const int lrow = lane & 15;   // row (A) / col (B) within 16
    const int lk   = lane >> 4;   // K half selector

    const int m0 = blockIdx.x * 256 + (wave >> 1) * 64;
    const int n0 = blockIdx.y * 128 + (wave & 1) * 64;

    const unsigned short* aRow[4];
#pragma unroll
    for (int i = 0; i < 4; ++i)
        aRow[i] = A + (size_t)(m0 + i * 16 + lrow) * DIMN;

    const unsigned short* bRow[4];
#pragma unroll
    for (int j = 0; j < 4; ++j)
        bRow[j] = W + (size_t)(n0 + j * 16 + lrow) * DIMN;

    v8f acc[4][4];
#pragma unroll
    for (int i = 0; i < 4; ++i)
#pragma unroll
        for (int j = 0; j < 4; ++j) {
            v8f z = {};
            acc[i][j] = z;
        }

    const int koff0 = lk * 8;        // elements 0..7  : K = k0 + lk*8 + 0..7
    const int koff1 = 16 + lk * 8;   // elements 8..15 : K = k0 + 16 + lk*8 + 0..7

    for (int k0 = 0; k0 < DIMN; k0 += 32) {
        Frag a[4], b[4];
#pragma unroll
        for (int i = 0; i < 4; ++i) {
            a[i].q[0] = *reinterpret_cast<const uint4*>(aRow[i] + k0 + koff0);
            a[i].q[1] = *reinterpret_cast<const uint4*>(aRow[i] + k0 + koff1);
        }
#pragma unroll
        for (int j = 0; j < 4; ++j) {
            b[j].q[0] = *reinterpret_cast<const uint4*>(bRow[j] + k0 + koff0);
            b[j].q[1] = *reinterpret_cast<const uint4*>(bRow[j] + k0 + koff1);
        }
#pragma unroll
        for (int i = 0; i < 4; ++i)
#pragma unroll
            for (int j = 0; j < 4; ++j)
                acc[i][j] = __builtin_amdgcn_wmma_f32_16x16x32_bf16(
                    false, a[i].v, false, b[j].v, (short)0, acc[i][j], false, false);
    }

    // Epilogue. C/D layout: lane l, vgpr r -> M = (l>>4)*8 + r, N = (l&15)
#pragma unroll
    for (int j = 0; j < 4; ++j) {
        const int col = n0 + j * 16 + lrow;
        const float bv = bias[col];
        const float sc = (MODE == 2) ? aux[col] : 0.0f;
#pragma unroll
        for (int i = 0; i < 4; ++i) {
#pragma unroll
            for (int r = 0; r < 8; ++r) {
                const int row = m0 + i * 16 + lk * 8 + r;
                const size_t idx = (size_t)row * DIMN + col;
                float v = acc[i][j][r] + bv;
                if (MODE == 0) {
                    outF[idx] = v;
                } else if (MODE == 1) {
                    outBF[idx] = f2bf(v);
                } else if (MODE == 2) {
                    outF[idx] = tanhf(v) * sc;
                } else {
                    outF[idx] = v + aux[idx];
                }
            }
        }
    }
}

// ---------------------------------------------------------------------------
// Phasor bind + per-chunk cumulative sum + retrieve.
// One thread per feature column; serial scan over the 64 chunk positions.
// ret may alias kp (per-thread read-before-write, no cross-thread sharing):
// kp/ret intentionally NOT __restrict__.
// ---------------------------------------------------------------------------
__global__ __launch_bounds__(256)
void k_bind_cumsum(const float* __restrict__ val,
                   const float* kp,
                   const float* __restrict__ qp,
                   float* ret)
{
    const int d  = blockIdx.x * 256 + threadIdx.x;   // 0..DIMN-1
    const int bc = blockIdx.y;                       // flat (batch, chunk): 0..MROWS/CHUNKN-1
    const size_t base = (size_t)bc * CHUNKN * DIMN + d;
    const float inv = 0.03125f;                      // 1/sqrt(1024)
    float cr = 0.0f, ci = 0.0f;
    for (int t = 0; t < CHUNKN; ++t) {
        const size_t idx = base + (size_t)t * DIMN;
        // Prefetch a few rows ahead of the serial scan dependency chain.
        if (t + 8 < CHUNKN) {
            const size_t pidx = idx + (size_t)8 * DIMN;
            __builtin_prefetch(&val[pidx], 0, 3);
            __builtin_prefetch(&kp[pidx], 0, 3);
            __builtin_prefetch(&qp[pidx], 0, 3);
        }
        const float v = val[idx];
        const float p = kp[idx];
        const float q = qp[idx];
        float sp, cp, sq, cq;
        __sincosf(p, &sp, &cp);
        __sincosf(q, &sq, &cq);
        cr += v * cp;
        ci += v * sp;
        ret[idx] = (cr * cq + ci * sq) * inv;
    }
}

// ---------------------------------------------------------------------------
// LayerNorm over last dim (1024), output bf16 for the final WMMA GEMM.
// One 256-thread block per row; each thread owns 4 columns.
// ---------------------------------------------------------------------------
__global__ __launch_bounds__(256)
void k_layernorm(const float* __restrict__ in,
                 const float* __restrict__ g,
                 const float* __restrict__ b,
                 unsigned short* __restrict__ out)
{
    __shared__ float s1[256];
    __shared__ float s2[256];
    const int row = blockIdx.x;
    const float* x = in + (size_t)row * DIMN;
    float v[4];
    float lsum = 0.0f, lsq = 0.0f;
#pragma unroll
    for (int p = 0; p < 4; ++p) {
        v[p] = x[threadIdx.x + p * 256];
        lsum += v[p];
        lsq  += v[p] * v[p];
    }
    s1[threadIdx.x] = lsum;
    s2[threadIdx.x] = lsq;
    __syncthreads();
    for (int off = 128; off > 0; off >>= 1) {
        if (threadIdx.x < off) {
            s1[threadIdx.x] += s1[threadIdx.x + off];
            s2[threadIdx.x] += s2[threadIdx.x + off];
        }
        __syncthreads();
    }
    const float mu  = s1[0] * (1.0f / DIMN);
    const float var = s2[0] * (1.0f / DIMN) - mu * mu;
    const float rs  = rsqrtf(var + 1e-5f);
    unsigned short* o = out + (size_t)row * DIMN;
#pragma unroll
    for (int p = 0; p < 4; ++p) {
        const int c = threadIdx.x + p * 256;
        o[c] = f2bf((v[p] - mu) * rs * g[c] + b[c]);
    }
}

// ---------------------------------------------------------------------------
// Launcher
// ---------------------------------------------------------------------------
extern "C" void kernel_launch(void* const* d_in, const int* in_sizes, int n_in,
                              void* d_out, int out_size, void* d_ws, size_t ws_size,
                              hipStream_t stream)
{
    const float* x   = (const float*)d_in[0];
    const float* Wk  = (const float*)d_in[1];
    const float* bk  = (const float*)d_in[2];
    const float* Wv  = (const float*)d_in[3];
    const float* bv  = (const float*)d_in[4];
    const float* Wq  = (const float*)d_in[5];
    const float* bq  = (const float*)d_in[6];
    const float* Wkp = (const float*)d_in[7];
    const float* bkp = (const float*)d_in[8];
    const float* Wqp = (const float*)d_in[9];
    const float* bqp = (const float*)d_in[10];
    const float* ps  = (const float*)d_in[11];
    const float* lng = (const float*)d_in[12];
    const float* lnb = (const float*)d_in[13];
    const float* Wo  = (const float*)d_in[14];
    const float* bo  = (const float*)d_in[15];

    // Workspace layout (bytes)
    const size_t SZ_XBF = (size_t)MROWS * DIMN * 2;   // 32 MiB
    const size_t SZ_WBF = (size_t)DIMN * DIMN * 2;    //  2 MiB
    const size_t SZ_F32 = (size_t)MROWS * DIMN * 4;   // 64 MiB

    char* ws = (char*)d_ws;
    size_t off = 0;
    unsigned short* xbf    = (unsigned short*)(ws + off); off += SZ_XBF;
    unsigned short* wkbf   = (unsigned short*)(ws + off); off += SZ_WBF;
    unsigned short* wvbf   = (unsigned short*)(ws + off); off += SZ_WBF;
    unsigned short* wqbf   = (unsigned short*)(ws + off); off += SZ_WBF;
    unsigned short* wkpbf  = (unsigned short*)(ws + off); off += SZ_WBF;
    unsigned short* wqpbf  = (unsigned short*)(ws + off); off += SZ_WBF;
    unsigned short* wobf   = (unsigned short*)(ws + off); off += SZ_WBF;
    unsigned short* keybf  = (unsigned short*)(ws + off); off += SZ_XBF;
    unsigned short* qrybf  = (unsigned short*)(ws + off); off += SZ_XBF;
    float*          valf   = (float*)(ws + off);          off += SZ_F32;
    float*          kpf    = (float*)(ws + off);          off += SZ_F32;
    float*          qpf    = (float*)(ws + off);          off += SZ_F32;
    unsigned short* normbf = (unsigned short*)(ws + off); off += SZ_XBF;
    float*          retf   = kpf;   // retrieved aliases key_phase (safe, see kernel)

    // 1) Convert activations and weights to bf16
    k_f32_to_bf16<<<(MROWS * DIMN) / 256, 256, 0, stream>>>(x, xbf, MROWS * DIMN);
    k_f32_to_bf16<<<(DIMN * DIMN) / 256, 256, 0, stream>>>(Wk,  wkbf,  DIMN * DIMN);
    k_f32_to_bf16<<<(DIMN * DIMN) / 256, 256, 0, stream>>>(Wv,  wvbf,  DIMN * DIMN);
    k_f32_to_bf16<<<(DIMN * DIMN) / 256, 256, 0, stream>>>(Wq,  wqbf,  DIMN * DIMN);
    k_f32_to_bf16<<<(DIMN * DIMN) / 256, 256, 0, stream>>>(Wkp, wkpbf, DIMN * DIMN);
    k_f32_to_bf16<<<(DIMN * DIMN) / 256, 256, 0, stream>>>(Wqp, wqpbf, DIMN * DIMN);
    k_f32_to_bf16<<<(DIMN * DIMN) / 256, 256, 0, stream>>>(Wo,  wobf,  DIMN * DIMN);

    const dim3 gg(MROWS / 256, DIMN / 128);
    const dim3 gb(256);

    // 2) key/value/query projections
    k_gemm<1><<<gg, gb, 0, stream>>>(xbf, wkbf, bk, nullptr, nullptr, keybf);
    k_gemm<0><<<gg, gb, 0, stream>>>(xbf, wvbf, bv, nullptr, valf, nullptr);
    k_gemm<1><<<gg, gb, 0, stream>>>(xbf, wqbf, bq, nullptr, nullptr, qrybf);

    // 3) phase projections with fused tanh * phase_scale
    k_gemm<2><<<gg, gb, 0, stream>>>(keybf, wkpbf, bkp, ps, kpf, nullptr);
    k_gemm<2><<<gg, gb, 0, stream>>>(qrybf, wqpbf, bqp, ps, qpf, nullptr);

    // 4) phasor bind + chunked cumsum + retrieve (scaled by 1/sqrt(dim))
    k_bind_cumsum<<<dim3(DIMN / 256, MROWS / CHUNKN), 256, 0, stream>>>(valf, kpf, qpf, retf);

    // 5) LayerNorm -> bf16
    k_layernorm<<<MROWS, 256, 0, stream>>>(retf, lng, lnb, normbf);

    // 6) output projection + residual
    k_gemm<3><<<gg, gb, 0, stream>>>(normbf, wobf, bo, x, (float*)d_out, nullptr);
}
